// MultiheadedAttention_13649406066792
// MI455X (gfx1250) — matrix operand order repaired and checked by reading, not compiled, not run on hardware
//
#include <hip/hip_runtime.h>
#include <hip/hip_bf16.h>

// ---------------------------------------------------------------------------
// MHA forward for MI455X (gfx1250, wave32, WMMA bf16 16x16x32, f32 accum)
// B=2, S=2048, D=1024, H=16, DK=64
// Data movement: GLOBAL_LOAD_ASYNC_TO_LDS_B128 (ASYNCcnt) for all LDS staging.
// ---------------------------------------------------------------------------

constexpr int Bc  = 2;
constexpr int Sc  = 2048;
constexpr int Dc  = 1024;
constexpr int Hc  = 16;
constexpr int DKc = 64;
constexpr int BSc = Bc * Sc;                  // 4096 tokens
constexpr size_t BSD = (size_t)BSc * Dc;      // 4,194,304
constexpr size_t DD  = (size_t)Dc * Dc;       // 1,048,576

typedef __attribute__((ext_vector_type(4)))  __bf16 v4bf;
typedef __attribute__((ext_vector_type(8)))  __bf16 v8bf;
typedef __attribute__((ext_vector_type(16))) __bf16 v16bf;
typedef __attribute__((ext_vector_type(8)))  float  v8f;

// fp32 -> bf16, round-to-nearest-even
__device__ __forceinline__ __bf16 f2bf(float f) {
    union { unsigned short u; __bf16 b; } cv;
    unsigned int x = __float_as_uint(f);
    cv.u = (unsigned short)((x + 0x7FFFu + ((x >> 16) & 1u)) >> 16);
    return cv.b;
}

// Build a 16-element bf16 fragment from two contiguous 8-element (16B) chunks.
__device__ __forceinline__ v16bf frag2(const __bf16* p0, const __bf16* p1) {
    v8bf lo = *(const v8bf*)p0;
    v8bf hi = *(const v8bf*)p1;
    return __builtin_shufflevector(lo, hi, 0,1,2,3,4,5,6,7,8,9,10,11,12,13,14,15);
}

// 16B global -> LDS via the CDNA5 async path (no VGPR round-trip, ASYNCcnt).
// Generic pointers to __shared__ truncate to the wave-relative LDS byte offset
// (low 32 bits of the flat address), which is what the VDST field takes.
__device__ __forceinline__ void async_copy16(void* lds_dst, const void* gsrc) {
    unsigned lofs = (unsigned)(unsigned long long)lds_dst;
    unsigned long long ga = (unsigned long long)gsrc;
    asm volatile("global_load_async_to_lds_b128 %0, %1, off"
                 :: "v"(lofs), "v"(ga) : "memory");
}
__device__ __forceinline__ void async_wait0() {
    asm volatile("s_wait_asynccnt 0" ::: "memory");
}

// ---------------------------------------------------------------------------
// fp32 -> bf16 conversion (vec4)
// ---------------------------------------------------------------------------
__global__ __launch_bounds__(256) void cvt_bf16(const float* __restrict__ s,
                                                __bf16* __restrict__ d, int n4) {
    int i = blockIdx.x * blockDim.x + threadIdx.x;
    if (i < n4) {
        float4 f = ((const float4*)s)[i];
        v4bf o;
        o[0] = f2bf(f.x); o[1] = f2bf(f.y); o[2] = f2bf(f.z); o[3] = f2bf(f.w);
        ((v4bf*)d)[i] = o;   // 8-byte store
    }
}

// ---------------------------------------------------------------------------
// GEMM: Y[t,n] = sum_k X[t,k] * W[n,k] + bias[n]   (torch Linear: x @ W^T + b)
// Block tile 64(M) x 128(N), K-step 64. 256 threads = 8 waves, 32x32 per wave.
// 8 WMMAs per wave between barriers; tiles staged with async LDS loads.
// MODE 0: f32 out [t*D + n]                      (final projection)
// MODE 1: bf16 out head-major   [B,H,S,DK]       (q, k)
// MODE 2: bf16 out head-transp. [B,H,DK,S]       (v -> ready-made B-frags for P.V)
// ---------------------------------------------------------------------------
template <int MODE>
__global__ __launch_bounds__(256) void gemm_bf16(const __bf16* __restrict__ X,
                                                 const __bf16* __restrict__ W,
                                                 const float*  __restrict__ bias,
                                                 __bf16* __restrict__ Ybf,
                                                 float*  __restrict__ Yf) {
    __shared__ __bf16 As[64][72];     // 64 K + 8 pad (row = 144B, 16B aligned)
    __shared__ __bf16 Bs[128][72];

    const int tid  = threadIdx.x;
    const int wave = tid >> 5;
    const int lane = tid & 31;
    const int l16  = lane & 15;
    const int lh   = lane >> 4;           // 0/1 lane half

    const int m0 = blockIdx.x * 64;       // token tile
    const int n0 = blockIdx.y * 128;      // feature tile
    const int wm = (wave & 1) * 32;
    const int wn = (wave >> 1) * 32;

    v8f c[2][2] = {};

    for (int k0 = 0; k0 < Dc; k0 += 64) {
        __syncthreads();
        // A tile: 64 rows x 64 -> 512 16B chunks, 2 per thread (async to LDS)
#pragma unroll
        for (int i = 0; i < 2; ++i) {
            int chunk = tid + i * 256;
            int r  = chunk >> 3;
            int kc = (chunk & 7) * 8;
            async_copy16(&As[r][kc], &X[(size_t)(m0 + r) * Dc + k0 + kc]);
        }
        // B tile: 128 rows x 64 -> 1024 chunks, 4 per thread
#pragma unroll
        for (int i = 0; i < 4; ++i) {
            int chunk = tid + i * 256;
            int r  = chunk >> 3;
            int kc = (chunk & 7) * 8;
            async_copy16(&Bs[r][kc], &W[(size_t)(n0 + r) * Dc + k0 + kc]);
        }
        async_wait0();
        __syncthreads();

#pragma unroll
        for (int kk = 0; kk < 2; ++kk) {
            v16bf a[2], b[2];
#pragma unroll
            for (int mi = 0; mi < 2; ++mi) {  // A frag: K = {lh*8..+7, 16+lh*8..+7}
                const __bf16* p = &As[wm + mi * 16 + l16][kk * 32];
                a[mi] = frag2(p + lh * 8, p + 16 + lh * 8);
            }
#pragma unroll
            for (int nj = 0; nj < 2; ++nj) {  // B frag: K = {lh*8..+7, 16+lh*8..+7}
                const __bf16* p = &Bs[wn + nj * 16 + l16][kk * 32];
                b[nj] = frag2(p + lh * 8, p + 16 + lh * 8);
            }
#pragma unroll
            for (int mi = 0; mi < 2; ++mi)
#pragma unroll
                for (int nj = 0; nj < 2; ++nj)
                    c[mi][nj] = __builtin_amdgcn_wmma_f32_16x16x32_bf16(
                        false, a[mi], false, b[nj], (short)0, c[mi][nj], false, false);
        }
    }

    // Epilogue: C layout — lane holds col n=(lane&15); VGPR r = row r (+8 for lanes>=16)
#pragma unroll
    for (int mi = 0; mi < 2; ++mi)
#pragma unroll
        for (int nj = 0; nj < 2; ++nj) {
            const int n  = n0 + wn + nj * 16 + l16;
            const float bn = bias[n];
#pragma unroll
            for (int r = 0; r < 8; ++r) {
                const int m = m0 + wm + mi * 16 + r + lh * 8;
                const float val = c[mi][nj][r] + bn;
                if (MODE == 0) {
                    Yf[(size_t)m * Dc + n] = val;
                } else {
                    const int bb = m >> 11, s = m & 2047;   // m / S, m % S
                    const int h  = n >> 6,  dk = n & 63;    // n / DK, n % DK
                    size_t idx;
                    if (MODE == 1) idx = ((size_t)(bb * Hc + h) * Sc + s) * DKc + dk;
                    else           idx = ((size_t)(bb * Hc + h) * DKc + dk) * Sc + s;
                    Ybf[idx] = f2bf(val);
                }
            }
        }
}

// ---------------------------------------------------------------------------
// Flash attention: one workgroup = 128 queries of one (b,h); 8 waves x 16 rows.
// q frags live in registers for the whole loop; 64-key K/V blocks staged into
// LDS with async loads and shared by all 8 waves. Online softmax with 16-lane
// shuffle reductions (row stats replicated across each lane half).
// ---------------------------------------------------------------------------
__global__ __launch_bounds__(256) void attn_fwd(const __bf16* __restrict__ qh,  // [B,H,S,DK]
                                                const __bf16* __restrict__ kh,  // [B,H,S,DK]
                                                const __bf16* __restrict__ vt,  // [B,H,DK,S]
                                                __bf16* __restrict__ attn) {    // [BS, D]
    __shared__ __bf16 Ks[64][72];       // [key][dk]
    __shared__ __bf16 Vt[64][72];       // [dk][key]
    __shared__ __bf16 Pb[8][16][72];    // per-wave P tile [row][key]

    const int tid  = threadIdx.x;
    const int wave = tid >> 5;
    const int lane = tid & 31;
    const int l16  = lane & 15;
    const int lh   = lane >> 4;

    const int q0 = blockIdx.x * 128;          // query tile
    const int bh = blockIdx.y;                // b*H + h
    const int h  = bh & (Hc - 1);
    const int bb = bh >> 4;

    // Preload this wave's 16 q rows as A-fragments (2 K-steps of 32 over DK=64)
    v16bf qa[2];
    {
        const __bf16* qr = qh + ((size_t)bh * Sc + q0 + wave * 16 + l16) * DKc;
#pragma unroll
        for (int kk = 0; kk < 2; ++kk)
            qa[kk] = frag2(qr + kk * 32 + lh * 8, qr + kk * 32 + 16 + lh * 8);
    }

    float mrun[8], lrun[8];
#pragma unroll
    for (int r = 0; r < 8; ++r) { mrun[r] = -1e30f; lrun[r] = 0.0f; }
    v8f o[4] = {};

    const float sscale = 0.125f;              // 1/sqrt(DK)

    for (int kb = 0; kb < Sc; kb += 64) {
        __syncthreads();
        // stage K (row-major) and V^T (row-major): 2x 512 chunks, async to LDS
#pragma unroll
        for (int i = 0; i < 2; ++i) {
            int chunk = tid + i * 256;
            int r  = chunk >> 3;              // 0..63
            int cc = (chunk & 7) * 8;         // 0..56
            async_copy16(&Ks[r][cc], &kh[((size_t)bh * Sc + kb + r) * DKc + cc]);
            async_copy16(&Vt[r][cc], &vt[((size_t)bh * DKc + r) * Sc + kb + cc]);
        }
        async_wait0();
        __syncthreads();

        // ---- S = (q . k^T) * scale : 16 x 64 per wave --------------------
        v8f sc[4] = {};
#pragma unroll
        for (int nj = 0; nj < 4; ++nj)
#pragma unroll
            for (int kk = 0; kk < 2; ++kk) {
                const __bf16* p = &Ks[nj * 16 + l16][kk * 32];
                v16bf bf = frag2(p + lh * 8, p + 16 + lh * 8);
                sc[nj] = __builtin_amdgcn_wmma_f32_16x16x32_bf16(
                    false, qa[kk], false, bf, (short)0, sc[nj], false, false);
            }
#pragma unroll
        for (int nj = 0; nj < 4; ++nj)
#pragma unroll
            for (int r = 0; r < 8; ++r) sc[nj][r] = sc[nj][r] * sscale;

        // ---- online softmax ---------------------------------------------
        float mnew[8], alpha[8];
#pragma unroll
        for (int r = 0; r < 8; ++r) {
            float v = fmaxf(fmaxf(sc[0][r], sc[1][r]), fmaxf(sc[2][r], sc[3][r]));
#pragma unroll
            for (int off = 1; off < 16; off <<= 1) v = fmaxf(v, __shfl_xor(v, off));
            mnew[r]  = fmaxf(mrun[r], v);
            alpha[r] = __expf(mrun[r] - mnew[r]);
        }
        float rsum[8];
#pragma unroll
        for (int r = 0; r < 8; ++r) rsum[r] = 0.0f;
#pragma unroll
        for (int nj = 0; nj < 4; ++nj)
#pragma unroll
            for (int r = 0; r < 8; ++r) {
                float p = __expf(sc[nj][r] - mnew[r]);
                sc[nj][r] = p;
                rsum[r] += p;
            }
#pragma unroll
        for (int r = 0; r < 8; ++r) {
            float v = rsum[r];
#pragma unroll
            for (int off = 1; off < 16; off <<= 1) v += __shfl_xor(v, off);
            lrun[r] = lrun[r] * alpha[r] + v;
            mrun[r] = mnew[r];
        }
#pragma unroll
        for (int dj = 0; dj < 4; ++dj)
#pragma unroll
            for (int r = 0; r < 8; ++r) o[dj][r] = o[dj][r] * alpha[r];

        // ---- C-layout -> A-layout for P via per-wave LDS staging ---------
#pragma unroll
        for (int nj = 0; nj < 4; ++nj)
#pragma unroll
            for (int r = 0; r < 8; ++r)
                Pb[wave][r + lh * 8][nj * 16 + l16] = f2bf(sc[nj][r]);
        asm volatile("s_wait_dscnt 0" ::: "memory");   // same-wave LDS RAW order

        v16bf pa[2];
#pragma unroll
        for (int kk = 0; kk < 2; ++kk) {
            const __bf16* p = &Pb[wave][l16][0];
            pa[kk] = frag2(p + kk * 32 + lh * 8, p + kk * 32 + 16 + lh * 8);
        }

        // ---- O += P . V  (B-frags straight out of transposed V tile) -----
#pragma unroll
        for (int dj = 0; dj < 4; ++dj)
#pragma unroll
            for (int kk = 0; kk < 2; ++kk) {
                const __bf16* p = &Vt[dj * 16 + l16][kk * 32];
                v16bf vf = frag2(p + lh * 8, p + 16 + lh * 8);
                o[dj] = __builtin_amdgcn_wmma_f32_16x16x32_bf16(
                    false, pa[kk], false, vf, (short)0, o[dj], false, false);
            }
    }

    // ---- normalize and write attn output [token, h*64+dk] ---------------
#pragma unroll
    for (int dj = 0; dj < 4; ++dj)
#pragma unroll
        for (int r = 0; r < 8; ++r) {
            const int s = q0 + wave * 16 + r + lh * 8;
            const int d = h * 64 + dj * 16 + l16;
            attn[((size_t)(bb * Sc + s)) * Dc + d] = f2bf(o[dj][r] / lrun[r]);
        }
}

// ---------------------------------------------------------------------------
extern "C" void kernel_launch(void* const* d_in, const int* in_sizes, int n_in,
                              void* d_out, int out_size, void* d_ws, size_t ws_size,
                              hipStream_t stream) {
    const float* Q  = (const float*)d_in[0];
    const float* K  = (const float*)d_in[1];
    const float* V  = (const float*)d_in[2];
    const float* Wq = (const float*)d_in[3];
    const float* bq = (const float*)d_in[4];
    const float* Wk = (const float*)d_in[5];
    const float* bk = (const float*)d_in[6];
    const float* Wv = (const float*)d_in[7];
    const float* bv = (const float*)d_in[8];
    const float* Wo = (const float*)d_in[9];
    const float* bo = (const float*)d_in[10];
    float* out = (float*)d_out;

    // Workspace layout (bf16): 7*BSD + 4*DD elements = 64 MB total
    __bf16* w   = (__bf16*)d_ws;
    __bf16* Xq  = w;
    __bf16* Xk  = Xq + BSD;
    __bf16* Xv  = Xk + BSD;
    __bf16* Wqb = Xv + BSD;
    __bf16* Wkb = Wqb + DD;
    __bf16* Wvb = Wkb + DD;
    __bf16* Wob = Wvb + DD;
    __bf16* qhd = Wob + DD;   // [B,H,S,DK]
    __bf16* khd = qhd + BSD;  // [B,H,S,DK]
    __bf16* vtd = khd + BSD;  // [B,H,DK,S]
    __bf16* atn = vtd + BSD;  // [BS, D]

    if (ws_size < (size_t)(7 * BSD + 4 * DD) * sizeof(__bf16)) return;

    const int n4x = (int)(BSD / 4), n4w = (int)(DD / 4);
    cvt_bf16<<<n4x / 256, 256, 0, stream>>>(Q,  Xq,  n4x);
    cvt_bf16<<<n4x / 256, 256, 0, stream>>>(K,  Xk,  n4x);
    cvt_bf16<<<n4x / 256, 256, 0, stream>>>(V,  Xv,  n4x);
    cvt_bf16<<<n4w / 256, 256, 0, stream>>>(Wq, Wqb, n4w);
    cvt_bf16<<<n4w / 256, 256, 0, stream>>>(Wk, Wkb, n4w);
    cvt_bf16<<<n4w / 256, 256, 0, stream>>>(Wv, Wvb, n4w);
    cvt_bf16<<<n4w / 256, 256, 0, stream>>>(Wo, Wob, n4w);

    dim3 gg(BSc / 64, Dc / 128), gb(256);
    gemm_bf16<1><<<gg, gb, 0, stream>>>(Xq, Wqb, bq, qhd, nullptr);
    gemm_bf16<1><<<gg, gb, 0, stream>>>(Xk, Wkb, bk, khd, nullptr);
    gemm_bf16<2><<<gg, gb, 0, stream>>>(Xv, Wvb, bv, vtd, nullptr);

    attn_fwd<<<dim3(Sc / 128, Bc * Hc), 256, 0, stream>>>(qhd, khd, vtd, atn);

    gemm_bf16<0><<<gg, gb, 0, stream>>>(atn, Wob, bo, nullptr, out);
}
